// MoE_4973572128970
// MI455X (gfx1250) — compile-verified, matrix operand order checked by reading, not run on hardware
//
#include <hip/hip_runtime.h>
#include <hip/hip_bf16.h>

#define T_TOK   8192
#define D_IN    1024
#define DFF     4096
#define NEXP    8
#define NROUTED 7

#define BM      32            // tokens per block
#define BN      256           // N tile (both phases)
#define FFCHUNK 512           // DFF slice held in LDS per block
#define KSTEP   32            // WMMA bf16 K
#define HS_STRIDE (FFCHUNK + 8)   // 520 elems -> 1040B row stride (16B multiple)

typedef __attribute__((ext_vector_type(16))) __bf16 v16bf;
typedef __attribute__((ext_vector_type(8)))  __bf16 v8bf;
typedef __attribute__((ext_vector_type(2)))  __bf16 v2bf;
typedef __attribute__((ext_vector_type(8)))  float  v8f;
typedef __attribute__((ext_vector_type(2)))  float  v2f;

// ---------------------------------------------------------------------------
// Fragment loader: 16-bit A/B layout (ISA 7.12.2).
// lane<16: M=lane,    K = [k0..k0+8) and [k0+16..k0+24)
// lane>=16:M=lane-16, K = [k0+8..k0+16) and [k0+24..k0+32)
// Tiles stored row-major [m][k]; B tiles staged transposed so same path works.
// ---------------------------------------------------------------------------
__device__ __forceinline__ v16bf load_frag16(const __bf16* base, int stride,
                                             int row, int k0, int lane) {
    const int m  = lane & 15;
    const int kk = (lane & 16) ? 8 : 0;
    const __bf16* p = base + (row + m) * stride + k0 + kk;
    v8bf lo = *reinterpret_cast<const v8bf*>(p);        // ds_load_b128
    v8bf hi = *reinterpret_cast<const v8bf*>(p + 16);   // ds_load_b128
    return __builtin_shufflevector(lo, hi, 0,1,2,3,4,5,6,7,8,9,10,11,12,13,14,15);
}

// Branchless erf (Abramowitz-Stegun 7.1.26): 1 v_exp_f32 + fused FMAs, no
// divergent exec-mask paths (libm erff -> VALU bloat).
__device__ __forceinline__ float gelu_fast(float v) {
    const float xs = v * 0.70710678118654752f;
    const float ax = __builtin_fabsf(xs);
    const float t  = 1.0f / (1.0f + 0.3275911f * ax);
    float y = 1.061405429f;
    y = y * t - 1.453152027f;
    y = y * t + 1.421413741f;
    y = y * t - 0.284496736f;
    y = y * t + 0.254829592f;
    const float e   = __expf(-ax * ax);
    const float erf = __builtin_copysignf(1.0f - y * t * e, xs);
    return 0.5f * v * (1.0f + erf);
}

// Packed f32x2 -> bf16x2: vector-convert form selects a single
// two-operand v_cvt_pk_bf16_f32 (the bit-shift form produced 7 VALU ops/pair).
__device__ __forceinline__ unsigned pack2bf(float a, float b) {
    v2f f; f.x = a; f.y = b;
    return __builtin_bit_cast(unsigned, __builtin_convertvector(f, v2bf));
}

// ---- W staging, split into (global->regs) and (regs->LDS) halves ---------
// Per thread: 2 micro-tiles of 4k x 4n. 8 global_load_b128, then
// 16 packed cvt + 8 ds_store_b64 (transposed to [n][k]).
__device__ __forceinline__ void load_W_regs(float4* r, const float* __restrict__ w,
                                            int ld, int kbase, int nbase, int tid,
                                            bool pf) {
    #pragma unroll
    for (int g = 0; g < 2; ++g) {
        const int G  = tid + g * 256;       // 0..511 micro-tiles
        const int kq = (G >> 6) * 4;        // 8 k-quads
        const int nq = (G & 63) * 4;        // 64 n-quads
        const float* p = w + (size_t)(kbase + kq) * ld + nbase + nq;
        r[g * 4 + 0] = *reinterpret_cast<const float4*>(p);
        r[g * 4 + 1] = *reinterpret_cast<const float4*>(p + ld);
        r[g * 4 + 2] = *reinterpret_cast<const float4*>(p + 2 * (size_t)ld);
        r[g * 4 + 3] = *reinterpret_cast<const float4*>(p + 3 * (size_t)ld);
    }
    if (pf)                                  // global_prefetch_b8, 2 steps ahead
        __builtin_prefetch(w + (size_t)(kbase + 2 * KSTEP + (tid >> 6) * 4) * ld +
                           nbase + (tid & 63) * 4, 0, 1);
}

__device__ __forceinline__ void store_W_lds(__bf16* dst, const float4* r, int tid) {
    #pragma unroll
    for (int g = 0; g < 2; ++g) {
        const int G  = tid + g * 256;
        const int kq = (G >> 6) * 4;
        const int nq = (G & 63) * 4;
        const float4 a = r[g*4+0], b = r[g*4+1], c = r[g*4+2], d = r[g*4+3];
        uint2 s;
        s.x = pack2bf(a.x, b.x); s.y = pack2bf(c.x, d.x);
        *reinterpret_cast<uint2*>(dst + (nq + 0) * KSTEP + kq) = s;
        s.x = pack2bf(a.y, b.y); s.y = pack2bf(c.y, d.y);
        *reinterpret_cast<uint2*>(dst + (nq + 1) * KSTEP + kq) = s;
        s.x = pack2bf(a.z, b.z); s.y = pack2bf(c.z, d.z);
        *reinterpret_cast<uint2*>(dst + (nq + 2) * KSTEP + kq) = s;
        s.x = pack2bf(a.w, b.w); s.y = pack2bf(c.w, d.w);
        *reinterpret_cast<uint2*>(dst + (nq + 3) * KSTEP + kq) = s;
    }
}

__device__ __forceinline__ void store_X_lds(__bf16* dst_off, float4 xv) {
    uint2 pk; pk.x = pack2bf(xv.x, xv.y); pk.y = pack2bf(xv.z, xv.w);
    *reinterpret_cast<uint2*>(dst_off) = pk;
}

// ---------------------------------------------------------------------------
// Kernel 1: zero results + expert counters
// ---------------------------------------------------------------------------
__global__ void moe_init_kernel(float* __restrict__ out, int* __restrict__ cnt) {
    int gid = blockIdx.x * blockDim.x + threadIdx.x;
    reinterpret_cast<float4*>(out)[gid] = make_float4(0.f, 0.f, 0.f, 0.f);
    if (blockIdx.x == 0 && threadIdx.x < 16) cnt[threadIdx.x] = 0;
}

// ---------------------------------------------------------------------------
// Kernel 2: router — one wave32 per token. Logits, fp32 softmax, top-2,
// gathered per-expert token lists + gate weights.
// ---------------------------------------------------------------------------
__global__ __launch_bounds__(256)
void moe_router_kernel(const float* __restrict__ x,
                       const float* __restrict__ rw,
                       float* __restrict__ out_logits,
                       float* __restrict__ out_sel,
                       int*   __restrict__ cnt,
                       int*   __restrict__ idxl,
                       float* __restrict__ gatel) {
    const int tid  = blockIdx.x * blockDim.x + threadIdx.x;
    const int t    = tid >> 5;     // token id
    const int lane = tid & 31;

    float s[NROUTED];
    #pragma unroll
    for (int e = 0; e < NROUTED; ++e) s[e] = 0.f;

    const float* xr = x + (size_t)t * D_IN;
    for (int j = 0; j < D_IN / 32; ++j) {
        const int k = lane + j * 32;
        const float xv = xr[k];
        #pragma unroll
        for (int e = 0; e < NROUTED; ++e) s[e] += xv * rw[e * D_IN + k];
    }
    #pragma unroll
    for (int e = 0; e < NROUTED; ++e) {
        for (int off = 16; off > 0; off >>= 1)
            s[e] += __shfl_xor(s[e], off, 32);
    }

    if (lane == 0) {
        float m = s[0];
        #pragma unroll
        for (int e = 1; e < NROUTED; ++e) m = fmaxf(m, s[e]);
        float p[NROUTED], sum = 0.f;
        #pragma unroll
        for (int e = 0; e < NROUTED; ++e) { p[e] = __expf(s[e] - m); sum += p[e]; }
        const float inv = 1.f / sum;
        #pragma unroll
        for (int e = 0; e < NROUTED; ++e) p[e] *= inv;

        int a1 = 0;
        #pragma unroll
        for (int e = 1; e < NROUTED; ++e) if (p[e] > p[a1]) a1 = e;
        int a2 = (a1 == 0) ? 1 : 0;
        #pragma unroll
        for (int e = 0; e < NROUTED; ++e) if (e != a1 && p[e] > p[a2]) a2 = e;

        #pragma unroll
        for (int e = 0; e < NROUTED; ++e) out_logits[(size_t)t * NROUTED + e] = s[e];
        out_sel[t * 2 + 0] = (float)a1;
        out_sel[t * 2 + 1] = (float)a2;

        int pos1 = atomicAdd(&cnt[1 + a1], 1);
        idxl[a1 * T_TOK + pos1]  = t;
        gatel[a1 * T_TOK + pos1] = p[a1];
        int pos2 = atomicAdd(&cnt[1 + a2], 1);
        idxl[a2 * T_TOK + pos2]  = t;
        gatel[a2 * T_TOK + pos2] = p[a2];
    }
}

// ---------------------------------------------------------------------------
// Kernel 3: fused gathered expert MLP over one FF chunk:
//   out[tok,:] += gate * gelu(X[tok,:] @ W1[:,chunk]) @ W2[chunk,:]
// grid = (DFF/FFCHUNK, T/BM, NEXP); 256 threads = 8 waves, 2(M)x4(N) wave
// grid, wave tile 16x64 (4 WMMA accumulators). Double-buffered LDS with
// register-pipelined staging: barrier -> global loads (k+1) -> frag loads +
// 4 back-to-back WMMA (k) -> cvt+store (k+1). One barrier per k-step.
// ---------------------------------------------------------------------------
__global__ __launch_bounds__(256)
void moe_expert_kernel(const float* __restrict__ x,
                       const float* __restrict__ w1,
                       const float* __restrict__ w2,
                       float* __restrict__ out,
                       const int*   __restrict__ cnt,
                       const int*   __restrict__ idxl,
                       const float* __restrict__ gatel) {
    const int e      = blockIdx.z;
    const int mbase  = blockIdx.y * BM;
    const int ffbase = blockIdx.x * FFCHUNK;
    const int n_e    = (e == 0) ? T_TOK : cnt[e];
    if (mbase >= n_e) return;                 // uniform early exit

    __shared__ __bf16 Xs[2][BM * KSTEP];      // A tiles  [m][k]     2x2 KB
    __shared__ __bf16 Wsb[2][BN * KSTEP];     // B tiles  [n][k]     2x16 KB
    __shared__ __bf16 Hs[BM * HS_STRIDE];     // gelu(H)  [m][ff]    33 KB
    __shared__ int    s_tok[BM];
    __shared__ float  s_gate[BM];

    const int tid  = threadIdx.x;
    const int lane = tid & 31;
    const int wave = tid >> 5;
    const int wm   = wave >> 2;               // 0..1 -> 16-row slab
    const int wn   = wave & 3;                // 0..3 -> 64-col slab
    const int nw   = wn * 64;

    if (tid < BM) {
        const int i = mbase + tid;
        int tok; float g;
        if (e == 0)        { tok = i; g = 1.0f; }
        else if (i < n_e)  { tok = idxl[(e - 1) * T_TOK + i];
                             g   = gatel[(e - 1) * T_TOK + i]; }
        else               { tok = 0; g = 0.0f; }   // tail: contributes 0
        s_tok[tid] = tok; s_gate[tid] = g;
    }
    __syncthreads();

    // Hoisted per-thread gather pointer + LDS offset for X staging
    const float* xsrc = x + (size_t)s_tok[tid >> 3] * D_IN + (tid & 7) * 4;
    const int    xoff = (tid >> 3) * KSTEP + (tid & 7) * 4;

    const float* w1e = w1 + (size_t)e * D_IN * DFF;
    const float* w2e = w2 + (size_t)e * DFF * D_IN;

    constexpr int NK1 = D_IN / KSTEP;         // 32
    constexpr int NK2 = FFCHUNK / KSTEP;      // 16

    // ---------------- Phase 1: Hs = gelu(X @ W1[:, chunk]) ----------------
    for (int ffsub = 0; ffsub < FFCHUNK / BN; ++ffsub) {   // 2 iters
        const int nbase = ffbase + ffsub * BN;
        v8f acc[4] = {};
        float4 wr[8]; float4 xr;

        // prologue: fill buffer 0
        xr = *reinterpret_cast<const float4*>(xsrc);
        load_W_regs(wr, w1e, DFF, 0, nbase, tid, true);
        store_X_lds(&Xs[0][xoff], xr);
        store_W_lds(Wsb[0], wr, tid);

        #pragma unroll 2
        for (int kstep = 0; kstep < NK1; ++kstep) {
            __syncthreads();
            const int  cur  = kstep & 1;
            const bool more = (kstep + 1) < NK1;
            if (more) {                        // issue global loads early
                xr = *reinterpret_cast<const float4*>(xsrc + (kstep + 1) * KSTEP);
                load_W_regs(wr, w1e, DFF, (kstep + 1) * KSTEP, nbase, tid,
                            (kstep + 2) < NK1);
            }
            const v16bf a = load_frag16(Xs[cur], KSTEP, wm * 16, 0, lane);
            v16bf b[4];
            #pragma unroll
            for (int t2 = 0; t2 < 4; ++t2)
                b[t2] = load_frag16(Wsb[cur], KSTEP, nw + t2 * 16, 0, lane);
            #pragma unroll
            for (int t2 = 0; t2 < 4; ++t2)
                acc[t2] = __builtin_amdgcn_wmma_f32_16x16x32_bf16(
                    false, a, false, b[t2], (short)0, acc[t2], false, false);
            if (more) {                        // drain regs -> LDS after WMMAs
                store_X_lds(&Xs[cur ^ 1][xoff], xr);
                store_W_lds(Wsb[cur ^ 1], wr, tid);
            }
        }

        // branchless gelu, store bf16 to Hs (C-layout: lane>=16 -> rows +8)
        const int rbase = wm * 16 + ((lane & 16) ? 8 : 0);
        #pragma unroll
        for (int t2 = 0; t2 < 4; ++t2) {
            const int col = ffsub * BN + nw + t2 * 16 + (lane & 15);
            #pragma unroll
            for (int v = 0; v < 8; ++v)
                Hs[(rbase + v) * HS_STRIDE + col] = (__bf16)gelu_fast(acc[t2][v]);
        }
    }

    // ------------- Phase 2: out += gate * (Hs @ W2[chunk, :]) -------------
    for (int dsub = 0; dsub < D_IN / BN; ++dsub) {         // 4 iters
        const int dbase = dsub * BN;
        v8f acc[4] = {};
        float4 wr[8];

        load_W_regs(wr, w2e, D_IN, ffbase, dbase, tid, true);
        store_W_lds(Wsb[0], wr, tid);

        #pragma unroll 2
        for (int kstep = 0; kstep < NK2; ++kstep) {
            __syncthreads();                  // also fences Hs writes on 1st iter
            const int  cur  = kstep & 1;
            const bool more = (kstep + 1) < NK2;
            if (more)
                load_W_regs(wr, w2e, D_IN, ffbase + (kstep + 1) * KSTEP, dbase,
                            tid, (kstep + 2) < NK2);
            const v16bf a = load_frag16(Hs, HS_STRIDE, wm * 16, kstep * KSTEP, lane);
            v16bf b[4];
            #pragma unroll
            for (int t2 = 0; t2 < 4; ++t2)
                b[t2] = load_frag16(Wsb[cur], KSTEP, nw + t2 * 16, 0, lane);
            #pragma unroll
            for (int t2 = 0; t2 < 4; ++t2)
                acc[t2] = __builtin_amdgcn_wmma_f32_16x16x32_bf16(
                    false, a, false, b[t2], (short)0, acc[t2], false, false);
            if (more)
                store_W_lds(Wsb[cur ^ 1], wr, tid);
        }

        // gate-scaled scatter-accumulate (global_atomic_add_f32)
        const int rbase = wm * 16 + ((lane & 16) ? 8 : 0);
        #pragma unroll
        for (int t2 = 0; t2 < 4; ++t2) {
            const int col = dbase + nw + t2 * 16 + (lane & 15);
            #pragma unroll
            for (int v = 0; v < 8; ++v) {
                const int r = rbase + v;
                atomicAdd(out + (size_t)s_tok[r] * D_IN + col, s_gate[r] * acc[t2][v]);
            }
        }
    }
}

// ---------------------------------------------------------------------------
extern "C" void kernel_launch(void* const* d_in, const int* in_sizes, int n_in,
                              void* d_out, int out_size, void* d_ws, size_t ws_size,
                              hipStream_t stream) {
    const float* x  = (const float*)d_in[0];
    const float* rw = (const float*)d_in[1];
    const float* w1 = (const float*)d_in[2];
    const float* w2 = (const float*)d_in[3];

    float* out        = (float*)d_out;                       // [T, D] results
    float* out_logits = out + (size_t)T_TOK * D_IN;          // [T, 7]
    float* out_sel    = out_logits + (size_t)T_TOK * NROUTED;// [T, 2]

    int*   cnt   = (int*)d_ws;                               // [16] (1..7 used)
    int*   idxl  = cnt + 16;                                 // [7][T]
    float* gatel = (float*)(idxl + NROUTED * T_TOK);         // [7][T]

    moe_init_kernel<<<(T_TOK * D_IN) / (256 * 4), 256, 0, stream>>>(out, cnt);
    moe_router_kernel<<<(T_TOK * 32) / 256, 256, 0, stream>>>(
        x, rw, out_logits, out_sel, cnt, idxl, gatel);

    dim3 grid(DFF / FFCHUNK, T_TOK / BM, NEXP);
    moe_expert_kernel<<<grid, 256, 0, stream>>>(x, w1, w2, out, cnt, idxl, gatel);
}
